// Decoder_31078383354011
// MI455X (gfx1250) — matrix operand order repaired and checked by reading, not compiled, not run on hardware
//
#include <hip/hip_runtime.h>
#include <hip/hip_bf16.h>

// ---------------------------------------------------------------------------
// 3-layer GRU (Keras reset_after), B=128, T=256, UNITS=1024, fp32 I/O.
// MI455X / gfx1250 (wave32, WMMA bf16 -> f32 accumulate):
//   per layer:
//     1) XW = x @ W + b0 : 32768 x 3072 x 1024 GEMM, 128x128 block tiles,
//        B slab double-buffered through LDS (8 waves share it), WMMA bf16.
//     2) persistent recurrent kernel: 64 blocks; block owns 16 hidden units;
//        its 48x1024 U-weight slab is hoisted into LDS once (fits easily in
//        the 320KB WGP LDS), so each of the 256 sequential steps only reads
//        the L2-resident h state from global. Atomic grid barrier per step.
// LDS staging uses gfx1250 async global->LDS (ASYNCcnt) when available.
// ---------------------------------------------------------------------------

#define LAYERS 3
#define UNITS  1024
#define BATCH  128
#define TSTEPS 256
#define GDIM   (3 * UNITS)       // 3072
#define KDIM   UNITS             // 1024
#define MROWS  (BATCH * TSTEPS)  // 32768

typedef __attribute__((ext_vector_type(16))) __bf16   v16bf;
typedef __attribute__((ext_vector_type(8)))  float    v8f;
typedef __attribute__((ext_vector_type(4)))  unsigned int v4u;
typedef __attribute__((ext_vector_type(4)))  int       v4i;

// ---- optional gfx1250 async global->LDS path --------------------------------
#if defined(__has_builtin)
#  if __has_builtin(__builtin_amdgcn_global_load_async_to_lds_b128)
#    define HAVE_ASYNC_LDS 1
#  endif
#endif
#ifndef HAVE_ASYNC_LDS
#  define HAVE_ASYNC_LDS 0
#endif

__device__ __forceinline__ void copy16(unsigned short* lds_dst,
                                       const unsigned short* gsrc) {
#if HAVE_ASYNC_LDS
  __builtin_amdgcn_global_load_async_to_lds_b128(
      (__attribute__((address_space(1))) v4i*)gsrc,
      (__attribute__((address_space(3))) v4i*)lds_dst, 0, 0);
#else
  *(v4u*)lds_dst = *(const v4u*)gsrc;
#endif
}

__device__ __forceinline__ void async_join() {
#if HAVE_ASYNC_LDS
#  if __has_builtin(__builtin_amdgcn_s_wait_asynccnt)
  __builtin_amdgcn_s_wait_asynccnt(0);
#  else
  asm volatile("s_wait_asynccnt 0x0" ::: "memory");
#  endif
#endif
}

// float -> bf16 (round to nearest even)
__device__ __forceinline__ unsigned short f2bf(float f) {
  unsigned int u = __float_as_uint(f);
  u += 0x7FFFu + ((u >> 16) & 1u);
  return (unsigned short)(u >> 16);
}

// A fragment 16x32 bf16: lane<16 -> row=lane, K = k0..k0+15;
// lane>=16 holds K = k0+16..k0+31.  p points at (row, kk + (lane>>4)*16).
__device__ __forceinline__ v16bf load_a32(const unsigned short* p) {
  union { v4u u[2]; v16bf v; } r;
  r.u[0] = *(const v4u*)p;
  r.u[1] = *(const v4u*)(p + 8);
  return r.v;
}

// B fragment 32x16 bf16 from K-major rows: lane%16 = column, lane<16 holds
// K=kk..kk+15, lane>=16 holds K=kk+16..kk+31 (contiguous 16 elements).
__device__ __forceinline__ v16bf load_b32(const unsigned short* p) {
  union { v4u u[2]; v16bf v; } r;
  r.u[0] = *(const v4u*)p;
  r.u[1] = *(const v4u*)(p + 8);
  return r.v;
}

__device__ __forceinline__ v8f wmma_bf16(v16bf a, v16bf b, v8f c) {
  return __builtin_amdgcn_wmma_f32_16x16x32_bf16(
      false, a, false, b, (short)0, c, false, false);
}

// ---------------------------------------------------------------------------
// Conversion kernels
// ---------------------------------------------------------------------------
__global__ __launch_bounds__(256) void k_convert_bf16(
    const float* __restrict__ src, unsigned short* __restrict__ dst, size_t n) {
  size_t stride = (size_t)gridDim.x * blockDim.x;
  for (size_t i = (size_t)blockIdx.x * blockDim.x + threadIdx.x; i < n; i += stride)
    dst[i] = f2bf(src[i]);
}

// W/U [L][K=1024][G=3072] fp32 -> [L][G=3072][K=1024] bf16 (K-major rows)
__global__ __launch_bounds__(256) void k_transpose_bf16(
    const float* __restrict__ src, unsigned short* __restrict__ dst) {
  const size_t per_layer = (size_t)GDIM * KDIM;
  const size_t total = (size_t)LAYERS * per_layer;
  size_t stride = (size_t)gridDim.x * blockDim.x;
  for (size_t i = (size_t)blockIdx.x * blockDim.x + threadIdx.x; i < total; i += stride) {
    size_t l = i / per_layer;
    size_t rem = i - l * per_layer;
    size_t n = rem / KDIM;
    size_t k = rem - n * KDIM;
    dst[i] = f2bf(src[l * per_layer + k * GDIM + n]);
  }
}

__global__ __launch_bounds__(256) void k_init_h(
    const float* __restrict__ init_states, float* __restrict__ hF,
    unsigned short* __restrict__ hB0, int layer, int* __restrict__ bar) {
  int idx = blockIdx.x * blockDim.x + threadIdx.x;
  if (idx == 0) *bar = 0;
  if (idx < BATCH * UNITS) {
    int b = idx / UNITS, u = idx % UNITS;
    float v = init_states[((size_t)b * UNITS + u) * LAYERS + layer];
    hF[idx] = v;
    hB0[idx] = f2bf(v);
  }
}

// ---------------------------------------------------------------------------
// GEMM: XW[m][g] = sum_k Xbf[m][k] * Wt[g][k] + bias0[g]
// 256 threads (8 waves), tile 128(M) x 128(N); B slab staged in LDS in
// 64-K stages, double buffered; padded row stride 72 (36 dwords -> 16 frag
// rows map to 16 distinct 4-bank groups, conflict-free ds_load_b128).
// ---------------------------------------------------------------------------
#define KSTAGE 64
#define NSTAGE (KDIM / KSTAGE)  // 16
#define BPAD   72

__global__ __launch_bounds__(256) void k_gemm_xw(
    const unsigned short* __restrict__ Xbf,   // [MROWS][KDIM]
    const unsigned short* __restrict__ Wt,    // [GDIM][KDIM]
    const float* __restrict__ bias0,          // [GDIM]
    float* __restrict__ XW) {                 // [MROWS][GDIM]
  __shared__ alignas(16) unsigned short ldsB[2][128][BPAD];
  const int tid  = threadIdx.x;
  const int lane = tid & 31;
  const int wave = tid >> 5;
  const int m0 = blockIdx.x * 128 + wave * 16;
  const int n0 = blockIdx.y * 128;
  const int col = lane & 15;

  v8f acc[8];
#pragma unroll
  for (int i = 0; i < 8; ++i) acc[i] = (v8f){};

  const unsigned short* aRow =
      Xbf + (size_t)(m0 + col) * KDIM + ((lane >> 4) << 4);

  // prologue: stage 0 (128 rows x 64 K = 1024 x 16B chunks)
  for (int c = tid; c < 1024; c += 256) {
    int row = c >> 3, off = (c & 7) << 3;
    copy16(&ldsB[0][row][off], Wt + (size_t)(n0 + row) * KDIM + off);
  }
  async_join();
  __syncthreads();

  for (int s = 0; s < NSTAGE; ++s) {
    if (s + 1 < NSTAGE) {
      const int kk0 = (s + 1) * KSTAGE;
      for (int c = tid; c < 1024; c += 256) {
        int row = c >> 3, off = (c & 7) << 3;
        copy16(&ldsB[(s + 1) & 1][row][off],
               Wt + (size_t)(n0 + row) * KDIM + kk0 + off);
      }
    }
    const unsigned short* bbase = &ldsB[s & 1][0][0];
#pragma unroll
    for (int kh = 0; kh < 2; ++kh) {
      const int kkl = kh * 32;
      v16bf a = load_a32(aRow + s * KSTAGE + kkl);
#pragma unroll
      for (int nt = 0; nt < 8; ++nt) {
        const unsigned short* bp =
            bbase + (nt * 16 + col) * BPAD + kkl + ((lane >> 4) << 4);
        acc[nt] = wmma_bf16(a, load_b32(bp), acc[nt]);
      }
    }
    async_join();
    __syncthreads();
  }

  // D layout: VGPR r -> M = m0 + r + 8*(lane>>4), N = n0 + nt*16 + col
  const int mbase = m0 + ((lane >> 4) << 3);
#pragma unroll
  for (int nt = 0; nt < 8; ++nt) {
    int g = n0 + nt * 16 + col;
    float bb = bias0[g];
#pragma unroll
    for (int r = 0; r < 8; ++r)
      XW[(size_t)(mbase + r) * GDIM + g] = acc[nt][r] + bb;
  }
}

// ---------------------------------------------------------------------------
// Persistent recurrent kernel. grid = 64 blocks x 256 threads.
// Block owns units [bi*16, bi*16+16) -> gate columns u, 1024+u, 2048+u.
// Its 48x1024 bf16 U slab lives in LDS (row stride 1032 -> 516 dwords,
// 516 mod 64 = 4 -> conflict-free frag reads). Wave w owns batch rows
// [16w,16w+16). Double-buffered bf16 h; atomic grid barrier per step.
// ---------------------------------------------------------------------------
#define UROWS 48
#define UPAD  (KDIM + 8)   // 1032

__global__ __launch_bounds__(256) void k_gru_recur(
    const unsigned short* __restrict__ Ut,    // [GDIM][KDIM] bf16 (layer l)
    const float* __restrict__ bias1,          // [GDIM]
    const float* __restrict__ XW,             // [MROWS][GDIM]
    float* __restrict__ hF,                   // [BATCH][UNITS] fp32
    unsigned short* __restrict__ hB0,         // bf16 h (even t read)
    unsigned short* __restrict__ hB1,         // bf16 h (odd t read)
    unsigned short* __restrict__ ynext,       // [BATCH][TSTEPS][UNITS] bf16
    float* __restrict__ yout,                 // fp32 out (last layer) or null
    float* __restrict__ states_out,           // [BATCH][UNITS][LAYERS]
    int layer, int* __restrict__ bar) {
  __shared__ alignas(16) unsigned short ldsU[UROWS][UPAD];
  const int tid  = threadIdx.x;
  const int lane = tid & 31;
  const int wave = tid >> 5;
  const int u0 = blockIdx.x * 16;
  const int m0 = wave * 16;
  const int nblocks = gridDim.x;
  const int col = lane & 15;

  // hoist U slab into LDS once: 48 rows x 1024 = 6144 x 16B chunks
  for (int c = tid; c < 6144; c += 256) {
    int row = c >> 7;               // 128 chunks per row
    int off = (c & 127) << 3;
    int g3 = row >> 4, cc = row & 15;
    copy16(&ldsU[row][off], Ut + (size_t)(g3 * UNITS + u0 + cc) * KDIM + off);
  }
  async_join();
  __syncthreads();

  const unsigned short* bRow[3];
#pragma unroll
  for (int g3 = 0; g3 < 3; ++g3)
    bRow[g3] = &ldsU[g3 * 16 + col][(lane >> 4) << 4];

  const int mbase = m0 + ((lane >> 4) << 3);
  const float bz = bias1[u0 + col];
  const float br = bias1[UNITS + u0 + col];
  const float bh = bias1[2 * UNITS + u0 + col];
  const size_t aoff = (size_t)(m0 + col) * KDIM + ((lane >> 4) << 4);

  for (int t = 0; t < TSTEPS; ++t) {
    const unsigned short* hRead = (t & 1) ? hB1 : hB0;
    unsigned short* hWrite      = (t & 1) ? hB0 : hB1;
    const unsigned short* aRow = hRead + aoff;

    v8f az = (v8f){}, ar = (v8f){}, ah = (v8f){};
    for (int kk = 0; kk < KDIM; kk += 32) {
      v16bf a  = load_a32(aRow + kk);
      v16bf b0 = load_b32(bRow[0] + kk);
      v16bf b1 = load_b32(bRow[1] + kk);
      v16bf b2 = load_b32(bRow[2] + kk);
      az = wmma_bf16(a, b0, az);
      ar = wmma_bf16(a, b1, ar);
      ah = wmma_bf16(a, b2, ah);
    }

    const int u = u0 + col;
#pragma unroll
    for (int r = 0; r < 8; ++r) {
      const int m = mbase + r;                       // batch index
      const size_t xwoff = ((size_t)m * TSTEPS + t) * GDIM;
      float xz = XW[xwoff + u];
      float xr = XW[xwoff + UNITS + u];
      float xh = XW[xwoff + 2 * UNITS + u];
      float hz = az[r] + bz, hr = ar[r] + br, hh = ah[r] + bh;
      float z  = 1.0f / (1.0f + __expf(-(xz + hz)));
      float rr = 1.0f / (1.0f + __expf(-(xr + hr)));
      float hhat = tanhf(xh + rr * hh);
      const size_t hidx = (size_t)m * UNITS + u;
      float hnew = z * hF[hidx] + (1.0f - z) * hhat;
      hF[hidx] = hnew;
      hWrite[hidx] = f2bf(hnew);
      const size_t yidx = ((size_t)m * TSTEPS + t) * UNITS + u;
      ynext[yidx] = f2bf(hnew);
      if (yout) yout[yidx] = hnew;
      if (t == TSTEPS - 1)
        states_out[((size_t)m * UNITS + u) * LAYERS + layer] = hnew;
    }

    // ---- device-wide barrier (monotonic counter) ----
    __threadfence();
    __syncthreads();
    if (threadIdx.x == 0) {
      atomicAdd(bar, 1);
      const int target = nblocks * (t + 1);
      while (__hip_atomic_load(bar, __ATOMIC_RELAXED, __HIP_MEMORY_SCOPE_AGENT) < target) {
        __builtin_amdgcn_s_sleep(1);
      }
    }
    __syncthreads();
    __threadfence();
  }
}

// ---------------------------------------------------------------------------
// Host launcher
// ---------------------------------------------------------------------------
extern "C" void kernel_launch(void* const* d_in, const int* in_sizes, int n_in,
                              void* d_out, int out_size, void* d_ws, size_t ws_size,
                              hipStream_t stream) {
  (void)in_sizes; (void)n_in; (void)out_size; (void)ws_size;
  const float* x_in  = (const float*)d_in[0];  // [B][T][U]
  const float* h0_in = (const float*)d_in[1];  // [B][U][L]
  const float* W_in  = (const float*)d_in[2];  // [L][U][3U]
  const float* U_in  = (const float*)d_in[3];  // [L][U][3U]
  const float* b_in  = (const float*)d_in[4];  // [L][2][3U]

  float* out_x      = (float*)d_out;
  float* out_states = out_x + (size_t)BATCH * TSTEPS * UNITS;

  char* ws = (char*)d_ws;
  size_t off = 0;
  auto carve = [&](size_t bytes) -> char* {
    char* p = ws + off;
    off = (off + bytes + 255) & ~(size_t)255;
    return p;
  };
  const size_t wsz = (size_t)LAYERS * GDIM * KDIM * sizeof(unsigned short);
  unsigned short* Wt  = (unsigned short*)carve(wsz);
  unsigned short* Ut  = (unsigned short*)carve(wsz);
  unsigned short* xA  = (unsigned short*)carve((size_t)MROWS * KDIM * sizeof(unsigned short));
  unsigned short* xB  = (unsigned short*)carve((size_t)MROWS * KDIM * sizeof(unsigned short));
  float*          XW  = (float*)carve((size_t)MROWS * GDIM * sizeof(float));
  float*          hF  = (float*)carve((size_t)BATCH * UNITS * sizeof(float));
  unsigned short* hB0 = (unsigned short*)carve((size_t)BATCH * UNITS * sizeof(unsigned short));
  unsigned short* hB1 = (unsigned short*)carve((size_t)BATCH * UNITS * sizeof(unsigned short));
  int*            bar = (int*)carve(256);

  k_convert_bf16<<<4096, 256, 0, stream>>>(x_in, xA, (size_t)MROWS * KDIM);
  k_transpose_bf16<<<4096, 256, 0, stream>>>(W_in, Wt);
  k_transpose_bf16<<<4096, 256, 0, stream>>>(U_in, Ut);

  unsigned short* xin  = xA;
  unsigned short* xout = xB;
  for (int l = 0; l < LAYERS; ++l) {
    const unsigned short* Wt_l = Wt + (size_t)l * GDIM * KDIM;
    const unsigned short* Ut_l = Ut + (size_t)l * GDIM * KDIM;
    const float* b0_l = b_in + (size_t)l * 2 * GDIM;
    const float* b1_l = b0_l + GDIM;

    dim3 ggrid(MROWS / 128, GDIM / 128);  // 256 x 24
    k_gemm_xw<<<ggrid, 256, 0, stream>>>(xin, Wt_l, b0_l, XW);

    k_init_h<<<(BATCH * UNITS + 255) / 256, 256, 0, stream>>>(h0_in, hF, hB0, l, bar);

    k_gru_recur<<<UNITS / 16, 256, 0, stream>>>(
        Ut_l, b1_l, XW, hF, hB0, hB1, xout,
        (l == LAYERS - 1) ? out_x : nullptr, out_states, l, bar);

    unsigned short* tmp = xin; xin = xout; xout = tmp;
  }
}
